// Speller_37074157699662
// MI455X (gfx1250) — matrix operand order repaired and checked by reading, not compiled, not run on hardware
//
#include <hip/hip_runtime.h>

#define BB 64
#define TT 1024
#define DD 512
#define HH 512
#define MM 256
#define CC 63
#define SS 64
#define KIH0 576   // 575 padded to multiple of 32

typedef __attribute__((ext_vector_type(16))) __bf16 v16bf;
typedef __attribute__((ext_vector_type(8)))  __bf16 v8bf;
typedef __attribute__((ext_vector_type(8)))  float  v8f;
typedef __attribute__((ext_vector_type(4)))  int    v4i;

#if defined(__gfx1250__) && defined(__has_builtin)
#if __has_builtin(__builtin_amdgcn_global_load_async_to_lds_b128)
#define USE_ASYNC_COPY 1
#else
#define USE_ASYNC_COPY 0
#endif
#else
#define USE_ASYNC_COPY 0
#endif

#if USE_ASYNC_COPY
typedef __attribute__((address_space(1))) v4i g_v4i;   // global v4i
typedef __attribute__((address_space(3))) v4i l_v4i;   // LDS v4i
#endif

__device__ __forceinline__ void async_wait_all()
{
#if defined(__gfx1250__)
#if __has_builtin(__builtin_amdgcn_s_wait_asynccnt)
    __builtin_amdgcn_s_wait_asynccnt(0);
#else
    asm volatile("s_wait_asynccnt 0" ::: "memory");
#endif
#endif
}

__device__ __forceinline__ float sigmoidf_(float x) { return 1.f / (1.f + __expf(-x)); }

// ---------------------------------------------------------------------------
// bf16 WMMA GEMM:  out[M x N] = act( A[M x K] @ W[N x K]^T + bias (+ out) )
// One wave computes one 16x16 tile. All calls use Ntiles % 4 == 0, so the 4
// waves of a block share one 16-row A tile -> stage it in LDS via the CDNA5
// async copy path (ASYNCcnt), then feed WMMA A-fragments from LDS (ds_load),
// B-fragments straight from global (weights are L2-resident).
// tiles = (M/16)*(N/16) must equal gridDim.x*4.
// ---------------------------------------------------------------------------
template <bool ACC, bool RELU, bool OUT_BF16>
__global__ void __launch_bounds__(128)
gemm_bf16_wmma(const __bf16* __restrict__ A, int lda,
               const __bf16* __restrict__ W, int ldw,
               float* __restrict__ outF, __bf16* __restrict__ outB, int ldo,
               int Ntiles, int K,
               const float* __restrict__ bias0, const float* __restrict__ bias1)
{
    __shared__ alignas(16) __bf16 sA[16 * 1024];   // A tile: 16 rows x K (K<=1024)

    const int w    = threadIdx.x >> 5;
    const int wave = blockIdx.x * 4 + w;
    const int tn   = wave % Ntiles;
    const int tmb  = (blockIdx.x * 4) / Ntiles;    // uniform across the block
    const int lane = threadIdx.x & 31;
    const int half = lane >> 4;                    // 0 or 1
    const int l16  = lane & 15;
    const int m0 = tmb * 16, n0 = tn * 16;

    // ---- cooperative stage of the A tile into LDS ----
    {
        const __bf16* Ab = A + (size_t)m0 * lda;
        const int kc = K >> 3;                     // 16B chunks per row
        const int chunks = 16 * kc;                // multiple of 128 for all K used
        for (int e = threadIdx.x; e < chunks; e += 128) {
            int row  = e / kc;
            int col8 = e - row * kc;
            const __bf16* gp = Ab + (size_t)row * lda + col8 * 8;
            __bf16* lp = sA + row * K + col8 * 8;
#if USE_ASYNC_COPY
            __builtin_amdgcn_global_load_async_to_lds_b128(
                (g_v4i*)gp, (l_v4i*)lp, 0, 0);
#else
            *(v8bf*)lp = *(const v8bf*)gp;
#endif
        }
#if USE_ASYNC_COPY
        async_wait_all();
#endif
        __syncthreads();
    }

    float bv = 0.f;
    if (bias0) bv += bias0[n0 + l16];
    if (bias1) bv += bias1[n0 + l16];

    v8f acc;
#pragma unroll
    for (int v = 0; v < 8; ++v) {
        float c = bv;
        if (ACC) c += outF[(size_t)(m0 + v + 8 * half) * ldo + n0 + l16];
        acc[v] = c;
    }

    // A fragment (from LDS): lane row = l16; chunks K=[half*8..+7], K=[16+half*8..+7]
    const __bf16* arow = sA + l16 * K + half * 8;
    // B fragment (from global): lane col = n0+l16 (= row of W); chunks K=[half*8..+7], K=[16+half*8..+7]
    const __bf16* brow = W + (size_t)(n0 + l16) * ldw + half * 8;

    for (int k = 0; k < K; k += 32) {
        v8bf alo = *(const v8bf*)(arow + k);
        v8bf ahi = *(const v8bf*)(arow + k + 16);
        v8bf blo = *(const v8bf*)(brow + k);
        v8bf bhi = *(const v8bf*)(brow + k + 16);
        v16bf af, bfm;
#pragma unroll
        for (int i = 0; i < 8; ++i) {
            af[i] = alo[i]; af[i + 8] = ahi[i];
            bfm[i] = blo[i]; bfm[i + 8] = bhi[i];
        }
        acc = __builtin_amdgcn_wmma_f32_16x16x32_bf16(
            /*neg_a=*/false, af, /*neg_b=*/false, bfm,
            /*c_mod=*/(short)0, acc, /*reuse_a=*/false, /*reuse_b=*/false);
    }

#pragma unroll
    for (int v = 0; v < 8; ++v) {
        float d = acc[v];
        if (RELU) d = fmaxf(d, 0.f);
        size_t idx = (size_t)(m0 + v + 8 * half) * ldo + n0 + l16;
        if (OUT_BF16) outB[idx] = (__bf16)d;
        else          outF[idx] = d;
    }
}

// ---------------------------------------------------------------------------
// LSTM elementwise cell: g = [i|f|g|o] (B x 4H), c in/out, h out (f32 + bf16)
// ---------------------------------------------------------------------------
__global__ void lstm_cell_kernel(const float* __restrict__ g,
                                 float* __restrict__ c,
                                 float* __restrict__ h,
                                 __bf16* __restrict__ hb)
{
    int idx = blockIdx.x * blockDim.x + threadIdx.x;   // BB*HH threads
    int b = idx >> 9;          // / 512
    int j = idx & (HH - 1);
    const float* gr = g + (size_t)b * 4 * HH;
    float i_ = gr[j], f_ = gr[j + HH], gg = gr[j + 2 * HH], o_ = gr[j + 3 * HH];
    float cn = sigmoidf_(f_) * c[idx] + sigmoidf_(i_) * tanhf(gg);
    float hn = sigmoidf_(o_) * tanhf(cn);
    c[idx] = cn;
    h[idx] = hn;
    hb[idx] = (__bf16)hn;
}

// energy[b][t] = dot(comp_dec[b][:], comp_listener[b][t][:])  (K = MM = 256)
__global__ void energy_kernel(const float* __restrict__ cd,
                              const __bf16* __restrict__ cl,
                              float* __restrict__ E)
{
    __shared__ float s_cd[MM];
    int b = blockIdx.x;
    int t = blockIdx.y * blockDim.x + threadIdx.x;
    if (threadIdx.x < MM) s_cd[threadIdx.x] = cd[b * MM + threadIdx.x];
    __syncthreads();
    const __bf16* row = cl + ((size_t)b * TT + t) * MM;
    float s = 0.f;
#pragma unroll 4
    for (int m = 0; m < MM; m += 8) {
        v8bf v = *(const v8bf*)(row + m);
#pragma unroll
        for (int i = 0; i < 8; ++i) s += s_cd[m + i] * (float)v[i];
    }
    E[b * TT + t] = s;
}

__global__ void reset_flag_kernel(int* z) { *z = 0; }

__global__ void zerocol_kernel(const float* __restrict__ E, int* __restrict__ z)
{
    int t = blockIdx.x * blockDim.x + threadIdx.x;
    if (t >= TT) return;
    for (int b = 0; b < BB; ++b)
        if (E[b * TT + t] != 0.f) return;
    *z = 1;
}

// Per-column-pair correlation r_j over the batch, j in [0, TT-2)
__global__ void colstats_kernel(const float* __restrict__ E,
                                float* __restrict__ r,
                                float* __restrict__ bad)
{
    int j = blockIdx.x * blockDim.x + threadIdx.x;
    if (j >= TT - 2) return;
    float sa = 0, sb = 0, saa = 0, sbb = 0, sab = 0;
    bool alleq = true;
    for (int b = 0; b < BB; ++b) {
        float a = E[b * TT + j], bb2 = E[b * TT + j + 1];
        sa += a; sb += bb2; saa += a * a; sbb += bb2 * bb2; sab += a * bb2;
        alleq = alleq && (a == bb2);
    }
    float ma = sa * (1.f / BB), mb = sb * (1.f / BB);
    float cab = sab - (float)BB * ma * mb;
    float caa = saa - (float)BB * ma * ma;
    float cbb = sbb - (float)BB * mb * mb;
    float rr = cab * rsqrtf(caa * cbb + 1e-12f);
    float det = 1.f - rr * rr;
    r[j]   = rr;
    bad[j] = (det < 0.01f || alleq) ? 1.f : 0.f;
}

// ef1 = pdf * coppdf, then row-softmax over T=1024; writes att slice of d_out
__global__ void __launch_bounds__(256)
attention_kernel(const float* __restrict__ E, const float* __restrict__ r,
                 const float* __restrict__ bad, const int* __restrict__ z,
                 float* __restrict__ att)
{
    __shared__ float red[256];
    const int b = blockIdx.x, tid = threadIdx.x;
    const bool usez = (*z) != 0;
    float vals[4];
#pragma unroll
    for (int i = 0; i < 4; ++i) {
        int t = i * 256 + tid;
        float e = E[b * TT + t];
        float x;
        if (usez) {
            x = e;
        } else {
            float pdf = __expf(-0.5f * e * e) * 0.3989422804014327f;
            float cop = 1.f;
            if (t >= 1 && t <= TT - 2) {
                int j = t - 1;
                float rr = r[j];
                float det_s = fmaxf(1.f - rr * rr, 1e-6f);
                float a = E[b * TT + t - 1];
                float quad = rr * rr * (a * a + e * e) - 2.f * rr * a * e;
                cop = __expf(-0.5f * quad / det_s) * rsqrtf(det_s);
                if (bad[j] != 0.f) cop = 10.f;
            }
            x = pdf * cop;
        }
        vals[i] = x;
    }
    float mx = fmaxf(fmaxf(vals[0], vals[1]), fmaxf(vals[2], vals[3]));
    red[tid] = mx; __syncthreads();
    for (int s2 = 128; s2 > 0; s2 >>= 1) {
        if (tid < s2) red[tid] = fmaxf(red[tid], red[tid + s2]);
        __syncthreads();
    }
    mx = red[0]; __syncthreads();
    float se = 0.f;
#pragma unroll
    for (int i = 0; i < 4; ++i) { vals[i] = __expf(vals[i] - mx); se += vals[i]; }
    red[tid] = se; __syncthreads();
    for (int s2 = 128; s2 > 0; s2 >>= 1) {
        if (tid < s2) red[tid] += red[tid + s2];
        __syncthreads();
    }
    float inv = 1.f / red[0];
#pragma unroll
    for (int i = 0; i < 4; ++i)
        att[(size_t)b * TT + i * 256 + tid] = vals[i] * inv;
}

// context[b][d] = sum_t att[b][t] * listener_bf16[b][t][d]
__global__ void __launch_bounds__(512)
context_kernel(const float* __restrict__ att, const __bf16* __restrict__ lf,
               float* __restrict__ ctx)
{
    __shared__ float a[TT];
    int b = blockIdx.x, d = threadIdx.x;
    for (int i = d; i < TT; i += 512) a[i] = att[(size_t)b * TT + i];
    __syncthreads();
    const __bf16* base = lf + (size_t)b * TT * DD + d;
    float s = 0.f;
    for (int t = 0; t < TT; ++t) {
        if ((t & 63) == 0 && t + 128 < TT)   // stream-ahead hint -> global_prefetch_b8
            __builtin_prefetch((const void*)(base + (size_t)(t + 128) * DD), 0, 0);
        s += a[t] * (float)base[(size_t)t * DD];
    }
    ctx[b * DD + d] = s;
}

// Build concat=[h1|ctx] (bf16, for char GEMM) and next rnn_in=[gt_t|ctx|pad]
__global__ void pack_kernel(const float* __restrict__ h1, const float* __restrict__ ctx,
                            const float* __restrict__ gt, int s,
                            __bf16* __restrict__ concat, __bf16* __restrict__ rnn_in)
{
    int b = blockIdx.x, i = threadIdx.x;   // 1024 threads
    concat[b * (2 * HH) + i] =
        (__bf16)((i < HH) ? h1[b * HH + i] : ctx[b * DD + (i - HH)]);
    if (i < KIH0) {
        float v;
        if (i < CC)              v = gt[((size_t)b * SS + s) * CC + i];
        else if (i < CC + DD)    v = ctx[b * DD + (i - CC)];
        else                     v = 0.f;
        rnn_in[b * KIH0 + i] = (__bf16)v;
    }
}

// init_in = [onehot(0) over C | listener[:,0,:] | pad]
__global__ void init_rnn_kernel(const float* __restrict__ lf, __bf16* __restrict__ rnn_in)
{
    int b = blockIdx.x, i = threadIdx.x;   // KIH0 threads
    float v;
    if (i == 0)              v = 1.f;
    else if (i < CC)         v = 0.f;
    else if (i < CC + DD)    v = lf[(size_t)b * TT * DD + (i - CC)];
    else                     v = 0.f;
    rnn_in[b * KIH0 + i] = (__bf16)v;
}

// log_softmax over CC=63 valid columns of logits (B x 64) -> preds slice
__global__ void logsoftmax_kernel(const float* __restrict__ logits,
                                  float* __restrict__ preds)
{
    __shared__ float red[64];
    int b = blockIdx.x, tid = threadIdx.x;   // 64 threads
    float v = (tid < CC) ? logits[b * 64 + tid] : -3.0e38f;
    red[tid] = v; __syncthreads();
    for (int s2 = 32; s2 > 0; s2 >>= 1) {
        if (tid < s2) red[tid] = fmaxf(red[tid], red[tid + s2]);
        __syncthreads();
    }
    float mx = red[0]; __syncthreads();
    float e = (tid < CC) ? __expf(v - mx) : 0.f;
    red[tid] = e; __syncthreads();
    for (int s2 = 32; s2 > 0; s2 >>= 1) {
        if (tid < s2) red[tid] += red[tid + s2];
        __syncthreads();
    }
    float ls = v - mx - __logf(red[0]);
    if (tid < CC) preds[(size_t)b * CC + tid] = ls;
}

// ------------------------- conversion / init helpers -----------------------
__global__ void f2bf_kernel(const float* __restrict__ src, __bf16* __restrict__ dst, size_t n)
{
    size_t i = (size_t)blockIdx.x * blockDim.x + threadIdx.x;
    size_t stride = (size_t)gridDim.x * blockDim.x;
    for (; i < n; i += stride) dst[i] = (__bf16)src[i];
}

__global__ void pad2d_kernel(const float* __restrict__ src, __bf16* __restrict__ dst,
                             int srows, int drows, int scols, int dcols)
{
    size_t n = (size_t)drows * dcols;
    size_t i = (size_t)blockIdx.x * blockDim.x + threadIdx.x;
    size_t stride = (size_t)gridDim.x * blockDim.x;
    for (; i < n; i += stride) {
        int rr = (int)(i / dcols), cc2 = (int)(i % dcols);
        float v = (rr < srows && cc2 < scols) ? src[(size_t)rr * scols + cc2] : 0.f;
        dst[i] = (__bf16)v;
    }
}

__global__ void padbias_kernel(const float* __restrict__ src, float* __restrict__ dst)
{
    int i = threadIdx.x;   // 64 threads
    dst[i] = (i < CC) ? src[i] : 0.f;
}

__global__ void zero_f32_kernel(float* p, size_t n)
{
    size_t i = (size_t)blockIdx.x * blockDim.x + threadIdx.x;
    size_t stride = (size_t)gridDim.x * blockDim.x;
    for (; i < n; i += stride) p[i] = 0.f;
}

__global__ void zero_bf16_kernel(__bf16* p, size_t n)
{
    size_t i = (size_t)blockIdx.x * blockDim.x + threadIdx.x;
    size_t stride = (size_t)gridDim.x * blockDim.x;
    for (; i < n; i += stride) p[i] = (__bf16)0.f;
}

// ===========================================================================
extern "C" void kernel_launch(void* const* d_in, const int* in_sizes, int n_in,
                              void* d_out, int out_size, void* d_ws, size_t ws_size,
                              hipStream_t stream)
{
    (void)in_sizes; (void)n_in; (void)out_size; (void)ws_size;

    const float* lf     = (const float*)d_in[0];
    const float* gt     = (const float*)d_in[1];
    const float* phi_w  = (const float*)d_in[2];
    const float* phi_b  = (const float*)d_in[3];
    const float* psi_w  = (const float*)d_in[4];
    const float* psi_b  = (const float*)d_in[5];
    const float* w_ih0  = (const float*)d_in[6];
    const float* w_hh0  = (const float*)d_in[7];
    const float* b_ih0  = (const float*)d_in[8];
    const float* b_hh0  = (const float*)d_in[9];
    const float* w_ih1  = (const float*)d_in[10];
    const float* w_hh1  = (const float*)d_in[11];
    const float* b_ih1  = (const float*)d_in[12];
    const float* b_hh1  = (const float*)d_in[13];
    const float* char_w = (const float*)d_in[14];
    const float* char_b = (const float*)d_in[15];
    float* out = (float*)d_out;

    // ---- workspace layout (256B aligned) ----
    char* ws = (char*)d_ws;
    size_t off = 0;
    auto alloc = [&](size_t bytes) -> char* {
        char* p = ws + off;
        off = (off + bytes + 255) & ~(size_t)255;
        return p;
    };
    __bf16* lf_bf     = (__bf16*)alloc(sizeof(__bf16) * (size_t)BB * TT * DD);
    __bf16* compL     = (__bf16*)alloc(sizeof(__bf16) * (size_t)BB * TT * MM);
    __bf16* wih0_bf   = (__bf16*)alloc(sizeof(__bf16) * 4 * HH * KIH0);
    __bf16* whh0_bf   = (__bf16*)alloc(sizeof(__bf16) * 4 * HH * HH);
    __bf16* wih1_bf   = (__bf16*)alloc(sizeof(__bf16) * 4 * HH * HH);
    __bf16* whh1_bf   = (__bf16*)alloc(sizeof(__bf16) * 4 * HH * HH);
    __bf16* phi_bf    = (__bf16*)alloc(sizeof(__bf16) * MM * DD);
    __bf16* psi_bf    = (__bf16*)alloc(sizeof(__bf16) * MM * DD);
    __bf16* char_bf   = (__bf16*)alloc(sizeof(__bf16) * 64 * 2 * HH);
    float*  char_b_pad= (float*) alloc(sizeof(float) * 64);
    float*  states    = (float*) alloc(sizeof(float) * 4 * BB * HH);
    float*  h0 = states, *c0 = states + BB * HH, *h1 = states + 2 * BB * HH, *c1 = states + 3 * BB * HH;
    __bf16* hb        = (__bf16*)alloc(sizeof(__bf16) * 2 * BB * HH);
    __bf16* h0b = hb, *h1b = hb + BB * HH;
    __bf16* rnn_bf    = (__bf16*)alloc(sizeof(__bf16) * BB * KIH0);
    __bf16* concat_bf = (__bf16*)alloc(sizeof(__bf16) * BB * 2 * HH);
    float*  g         = (float*) alloc(sizeof(float) * BB * 4 * HH);
    float*  comp_dec  = (float*) alloc(sizeof(float) * BB * MM);
    float*  energy    = (float*) alloc(sizeof(float) * BB * TT);
    float*  ctx       = (float*) alloc(sizeof(float) * BB * DD);
    float*  rbuf      = (float*) alloc(sizeof(float) * TT);
    float*  badbuf    = (float*) alloc(sizeof(float) * TT);
    int*    zflag     = (int*)   alloc(256);
    float*  logits    = (float*) alloc(sizeof(float) * BB * 64);

    // ---- one-time conversions / padding ----
    f2bf_kernel<<<8192, 256, 0, stream>>>(lf, lf_bf, (size_t)BB * TT * DD);
    pad2d_kernel<<<512, 256, 0, stream>>>(w_ih0, wih0_bf, 4 * HH, 4 * HH, 575, KIH0);
    pad2d_kernel<<<512, 256, 0, stream>>>(w_hh0, whh0_bf, 4 * HH, 4 * HH, HH, HH);
    pad2d_kernel<<<512, 256, 0, stream>>>(w_ih1, wih1_bf, 4 * HH, 4 * HH, HH, HH);
    pad2d_kernel<<<512, 256, 0, stream>>>(w_hh1, whh1_bf, 4 * HH, 4 * HH, HH, HH);
    pad2d_kernel<<<128, 256, 0, stream>>>(phi_w, phi_bf, MM, MM, DD, DD);
    pad2d_kernel<<<128, 256, 0, stream>>>(psi_w, psi_bf, MM, MM, DD, DD);
    pad2d_kernel<<<64, 256, 0, stream>>>(char_w, char_bf, CC, 64, 2 * HH, 2 * HH);
    padbias_kernel<<<1, 64, 0, stream>>>(char_b, char_b_pad);
    zero_f32_kernel<<<128, 256, 0, stream>>>(states, (size_t)4 * BB * HH);
    zero_bf16_kernel<<<64, 256, 0, stream>>>(hb, (size_t)2 * BB * HH);
    init_rnn_kernel<<<BB, KIH0, 0, stream>>>(lf, rnn_bf);

    // comp_listener = relu(listener @ psi_w^T + psi_b), bf16 out (65536 x 256)
    gemm_bf16_wmma<false, true, true><<<(BB * TT / 16) * (MM / 16) / 4, 128, 0, stream>>>(
        lf_bf, DD, psi_bf, DD, nullptr, compL, MM, MM / 16, DD, psi_b, nullptr);

    float* preds_base = out;
    float* atts_base  = out + (size_t)SS * BB * CC;

    for (int s = 0; s < SS; ++s) {
        // ---- LSTM layer 0 ----
        gemm_bf16_wmma<false, false, false><<<128, 128, 0, stream>>>(
            rnn_bf, KIH0, wih0_bf, KIH0, g, nullptr, 4 * HH, (4 * HH) / 16, KIH0, b_ih0, b_hh0);
        gemm_bf16_wmma<true, false, false><<<128, 128, 0, stream>>>(
            h0b, HH, whh0_bf, HH, g, nullptr, 4 * HH, (4 * HH) / 16, HH, nullptr, nullptr);
        lstm_cell_kernel<<<128, 256, 0, stream>>>(g, c0, h0, h0b);

        // ---- LSTM layer 1 ----
        gemm_bf16_wmma<false, false, false><<<128, 128, 0, stream>>>(
            h0b, HH, wih1_bf, HH, g, nullptr, 4 * HH, (4 * HH) / 16, HH, b_ih1, b_hh1);
        gemm_bf16_wmma<true, false, false><<<128, 128, 0, stream>>>(
            h1b, HH, whh1_bf, HH, g, nullptr, 4 * HH, (4 * HH) / 16, HH, nullptr, nullptr);
        lstm_cell_kernel<<<128, 256, 0, stream>>>(g, c1, h1, h1b);

        // ---- attention ----
        gemm_bf16_wmma<false, true, false><<<16, 128, 0, stream>>>(
            h1b, HH, phi_bf, HH, comp_dec, nullptr, MM, MM / 16, HH, phi_b, nullptr);
        energy_kernel<<<dim3(BB, TT / 256), 256, 0, stream>>>(comp_dec, compL, energy);
        reset_flag_kernel<<<1, 1, 0, stream>>>(zflag);
        zerocol_kernel<<<TT / 256, 256, 0, stream>>>(energy, zflag);
        colstats_kernel<<<TT / 256, 256, 0, stream>>>(energy, rbuf, badbuf);

        float* att_s = atts_base + (size_t)s * BB * TT;
        attention_kernel<<<BB, 256, 0, stream>>>(energy, rbuf, badbuf, zflag, att_s);
        context_kernel<<<BB, 512, 0, stream>>>(att_s, lf_bf, ctx);

        // ---- output head + next input ----
        pack_kernel<<<BB, 1024, 0, stream>>>(h1, ctx, gt, s, concat_bf, rnn_bf);
        gemm_bf16_wmma<false, false, false><<<4, 128, 0, stream>>>(
            concat_bf, 2 * HH, char_bf, 2 * HH, logits, nullptr, 64, 64 / 16, 2 * HH,
            char_b_pad, nullptr);
        logsoftmax_kernel<<<BB, 64, 0, stream>>>(logits, preds_base + (size_t)s * BB * CC);
    }
}
